// GCAE_34050500723280
// MI455X (gfx1250) — matrix operand order repaired and checked by reading, not compiled, not run on hardware
//
#include <hip/hip_runtime.h>
#include <hip/hip_bf16.h>

// ST-GCN autoencoder for MI455X (gfx1250, wave32, WMMA).
// Channel matmuls (GCN 1x1, 9-tap temporal conv) on v_wmma_f32_16x16x32_f16
// (M=16 out-ch, N=16 (t,v) positions, K=32 in-ch, fp32 accumulate).
// Activation tiles stored TRANSPOSED in LDS ([col][chan], chan stride padded
// to 72 halves = 144B) so each lane's B-operand fetch is one contiguous 32B
// run -> ds_load_b128, bank-spread. Adjacency contraction (inner dim 54),
// BN, residual projection, bilinear-T upsample are VALU. Working set lives
// in L2; d_ws holds four ping-pong activation buffers (~85MB).

typedef __attribute__((ext_vector_type(16))) _Float16 v16h;
typedef __attribute__((ext_vector_type(8)))  float    v8f;

#define NB    512
#define VJ    18
#define KADJ  3
#define CMAX  64
#define CPAD  72    // CMAX + 8 halves: 144B row stride, avoids bank aliasing
#define TVPAD 224   // max T*V = 12*18 = 216, rounded to 16
#define SLABR 48    // KADJ * 16 rows of GCN weights per workgroup

// A-operand (16x32 f16) K index for element e (0..15) of lane L, matching B:
// lanes 0-15 hold K=0..15, lanes 16-31 hold K=16..31.
__device__ __forceinline__ int wmma_ka(int e, int lane) {
  return ((lane & 16) ? 16 : 0) + e;
}
// B-operand (32x16 f16): lanes 0-15 hold K=0..15, lanes 16-31 hold K=16..31.
__device__ __forceinline__ int wmma_kb(int e, int lane) {
  return ((lane & 16) ? 16 : 0) + e;
}

// -------------------------------------------------------------------------
// data BN: h[n,c,t,v] = x * g[v*3+c]*inv + b[v*3+c]
__global__ void databn_kernel(const float* __restrict__ X,
                              const float* __restrict__ g,
                              const float* __restrict__ b,
                              float* __restrict__ O) {
  const float inv = rsqrtf(1.0f + 1e-5f);
  const int total = NB * 3 * 12 * VJ;
  for (int i = blockIdx.x * blockDim.x + threadIdx.x; i < total;
       i += gridDim.x * blockDim.x) {
    int v  = i % VJ;
    int c  = (i / (VJ * 12)) % 3;
    int ch = v * 3 + c;
    O[i] = X[i] * g[ch] * inv + b[ch];
  }
}

// -------------------------------------------------------------------------
// Fused GCN (1x1 conv to K*Cout) + adjacency contraction + BN1 + ReLU.
// One workgroup: one sample n, one 16-channel output slab.
__global__ __launch_bounds__(256) void gcn_agg_kernel(
    const float* __restrict__ X,     // (N, Cin, T, V)
    const float* __restrict__ W,     // (KADJ*Cout, Cin)
    const float* __restrict__ gb,    // (KADJ*Cout)
    const float* __restrict__ Aadj,  // (KADJ, V, V)
    const float* __restrict__ ei,    // (KADJ, V, V)
    const float* __restrict__ b1g, const float* __restrict__ b1b,  // (Cout)
    float* __restrict__ H,           // (N, Cout, T, V)
    int Cin, int Cout, int T) {
  __shared__ _Float16 sXt[TVPAD][CPAD];  // transposed acts: 31.5 KB
  __shared__ _Float16 sW[SLABR][CPAD];   //  6.8 KB
  __shared__ _Float16 sG[SLABR][TVPAD];  // 21.0 KB
  __shared__ float    sA[KADJ][VJ][VJ];  //  3.8 KB
  __shared__ float    sBias[SLABR];

  const int tid = threadIdx.x;
  const int n   = blockIdx.y;
  const int c0  = blockIdx.x * 16;
  const int TV  = T * VJ;
  const int TVp = ((TV + 15) >> 4) << 4;
  const int Cp  = (Cin <= 32) ? 32 : 64;
  const int ntiles = TVp >> 4;

  for (int i = tid; i < KADJ * VJ * VJ; i += 256)
    (&sA[0][0][0])[i] = Aadj[i] * ei[i];

  for (int i = tid; i < SLABR * Cp; i += 256) {
    int lr = i / Cp, c = i % Cp;
    int kk = lr >> 4, rc = lr & 15;
    float w = 0.0f;
    if ((c0 + rc) < Cout && c < Cin)
      w = W[(size_t)(kk * Cout + c0 + rc) * Cin + c];
    sW[lr][c] = (_Float16)w;
  }
  for (int lr = tid; lr < SLABR; lr += 256) {
    int kk = lr >> 4, rc = lr & 15;
    sBias[lr] = ((c0 + rc) < Cout) ? gb[kk * Cout + c0 + rc] : 0.0f;
  }
  // transposed store: global reads coalesced over col, LDS rows bank-spread
  for (int i = tid; i < Cp * TVp; i += 256) {
    int c = i / TVp, col = i % TVp;
    float v = 0.0f;
    if (c < Cin && col < TV) v = X[((size_t)n * Cin + c) * TV + col];
    sXt[col][c] = (_Float16)v;
  }
  __syncthreads();

  const int wave = tid >> 5, lane = tid & 31;
  const int m = lane & 15;
  const int mofs = (lane & 16) ? 8 : 0;
  for (int kk = 0; kk < KADJ; ++kk) {
    const _Float16* arow = &sW[kk * 16 + m][0];
    for (int jt = wave; jt < ntiles; jt += 8) {
      int col0 = jt << 4;
      const _Float16* bcol = &sXt[col0 + m][0];
      v8f acc = {0.f, 0.f, 0.f, 0.f, 0.f, 0.f, 0.f, 0.f};
      auto chunk = [&](int kc) {
        v16h a, b;
#pragma unroll
        for (int e = 0; e < 16; ++e) {
          a[e] = arow[kc + wmma_ka(e, lane)];
          b[e] = bcol[kc + wmma_kb(e, lane)];
        }
        acc = __builtin_amdgcn_wmma_f32_16x16x32_f16(false, a, false, b,
                                                     (short)0, acc, false, false);
      };
      chunk(0);
      if (Cp == 64) chunk(32);
#pragma unroll
      for (int r = 0; r < 8; ++r) {
        int row = kk * 16 + r + mofs;
        sG[row][col0 + m] = (_Float16)(acc[r] + sBias[row]);
      }
    }
  }
  __syncthreads();

  const float inv = rsqrtf(1.0f + 1e-5f);
  for (int i = tid; i < 16 * TV; i += 256) {
    int cl = i / TV, rem = i % TV;
    int t = rem / VJ, w = rem % VJ;
    int c = c0 + cl;
    if (c >= Cout) continue;
    float s = 0.0f;
#pragma unroll
    for (int kk = 0; kk < KADJ; ++kk)
      for (int v = 0; v < VJ; ++v)
        s += (float)sG[kk * 16 + cl][t * VJ + v] * sA[kk][v][w];
    float o = s * b1g[c] * inv + b1b[c];
    H[(((size_t)n * Cout + c) * T + t) * VJ + w] = fmaxf(o, 0.0f);
  }
}

// -------------------------------------------------------------------------
// 9-tap temporal conv (stride s) + bias + optional BN2 + residual + optional
// ReLU. One workgroup: one sample, one 16-output-channel slab.
__global__ __launch_bounds__(256) void tcn_kernel(
    const float* __restrict__ Hin,  // (N, C, Tin, V)
    const float* __restrict__ Wt,   // (C, C, 9, 1)
    const float* __restrict__ tb,   // (C)
    const float* __restrict__ b2g, const float* __restrict__ b2b,  // or null
    const float* __restrict__ Res,  // (N, C, Tout, V) or null
    float* __restrict__ Out,        // (N, C, Tout, V)
    int C, int Tin, int stride, int Tout, int relu) {
  __shared__ _Float16 sHt[TVPAD][CPAD];    // transposed acts: 31.5 KB
  __shared__ _Float16 sWt[9][16][CPAD];    // 20.25 KB

  const int tid = threadIdx.x;
  const int n  = blockIdx.y;
  const int o0 = blockIdx.x * 16;
  const int TVin  = Tin * VJ;
  const int TVinp = ((TVin + 15) >> 4) << 4;
  const int TVout = Tout * VJ;
  const int TVop  = ((TVout + 15) >> 4) << 4;
  const int ntiles = TVop >> 4;
  const int Cp = (C <= 32) ? 32 : 64;

  for (int i = tid; i < Cp * TVinp; i += 256) {
    int c = i / TVinp, col = i % TVinp;
    float v = 0.0f;
    if (c < C && col < TVin) v = Hin[((size_t)n * C + c) * TVin + col];
    sHt[col][c] = (_Float16)v;
  }
  for (int i = tid; i < 9 * 16 * Cp; i += 256) {
    int dt = i / (16 * Cp), rem = i % (16 * Cp);
    int ol = rem / Cp, c = rem % Cp;
    int o = o0 + ol;
    float w = 0.0f;
    if (o < C && c < C) w = Wt[((size_t)o * C + c) * 9 + dt];
    sWt[dt][ol][c] = (_Float16)w;
  }
  __syncthreads();

  const int wave = tid >> 5, lane = tid & 31;
  const int m = lane & 15;
  const int mofs = (lane & 16) ? 8 : 0;
  const float inv = rsqrtf(1.0f + 1e-5f);

  for (int tile = wave; tile < ntiles; tile += 8) {
    int col0 = tile << 4;
    int col  = col0 + m;
    bool cok = (col < TVout);
    int tp = col / VJ, v = col % VJ;
    v8f acc = {0.f, 0.f, 0.f, 0.f, 0.f, 0.f, 0.f, 0.f};
#pragma unroll
    for (int dt = 0; dt < 9; ++dt) {
      int tin = tp * stride + dt - 4;
      bool valid = cok && (tin >= 0) && (tin < Tin);
      int bidx = valid ? (tin * VJ + v) : 0;
      const _Float16* arow = &sWt[dt][m][0];
      const _Float16* bcol = &sHt[bidx][0];
      auto chunk = [&](int kc) {
        v16h a, b;
#pragma unroll
        for (int e = 0; e < 16; ++e) {
          a[e] = arow[kc + wmma_ka(e, lane)];
          _Float16 hv = bcol[kc + wmma_kb(e, lane)];
          b[e] = valid ? hv : (_Float16)0.0f;  // per-lane cndmask; EXEC stays full
        }
        acc = __builtin_amdgcn_wmma_f32_16x16x32_f16(false, a, false, b,
                                                     (short)0, acc, false, false);
      };
      chunk(0);
      if (Cp == 64) chunk(32);
    }
#pragma unroll
    for (int r = 0; r < 8; ++r) {
      int o = o0 + r + mofs;
      if (o < C && cok) {
        float val = acc[r] + tb[o];
        if (b2g) val = val * b2g[o] * inv + b2b[o];
        if (Res) val += Res[((size_t)n * C + o) * TVout + col];
        if (relu) val = fmaxf(val, 0.0f);
        Out[((size_t)n * C + o) * TVout + col] = val;
      }
    }
  }
}

// -------------------------------------------------------------------------
// Residual projection: R[n,o,t,v] = bn(res_w . x[:, :, t*stride, :] + res_b)
__global__ void resproj_kernel(const float* __restrict__ X,
                               const float* __restrict__ Wr,
                               const float* __restrict__ rb,
                               const float* __restrict__ g,
                               const float* __restrict__ bb,
                               float* __restrict__ R,
                               int Cin, int Cout, int Tin, int stride, int Tout) {
  const float inv = rsqrtf(1.0f + 1e-5f);
  const int total = NB * Cout * Tout * VJ;
  for (int i = blockIdx.x * blockDim.x + threadIdx.x; i < total;
       i += gridDim.x * blockDim.x) {
    int v = i % VJ;
    int t = (i / VJ) % Tout;
    int o = (i / (VJ * Tout)) % Cout;
    int n = i / (VJ * Tout * Cout);
    float s = rb[o];
    const float* xp = X + (size_t)n * Cin * Tin * VJ + (size_t)(t * stride) * VJ + v;
    for (int c = 0; c < Cin; ++c)
      s += Wr[(size_t)o * Cin + c] * xp[(size_t)c * Tin * VJ];
    R[i] = s * g[o] * inv + bb[o];
  }
}

// -------------------------------------------------------------------------
// Bilinear upsample along T (half-pixel centers, edge clamp).
__global__ void upsample_kernel(const float* __restrict__ In,
                                float* __restrict__ Out,
                                int C, int Tin, int sc) {
  const int Tout = Tin * sc;
  const int total = NB * C * Tout * VJ;
  for (int i = blockIdx.x * blockDim.x + threadIdx.x; i < total;
       i += gridDim.x * blockDim.x) {
    int v = i % VJ;
    int t = (i / VJ) % Tout;
    int c = (i / (VJ * Tout)) % C;
    int n = i / (VJ * Tout * C);
    float src = (t + 0.5f) / (float)sc - 0.5f;
    float fl = floorf(src);
    float w = src - fl;
    int t0 = (int)fl;
    int t0c = min(max(t0, 0), Tin - 1);
    int t1c = min(max(t0 + 1, 0), Tin - 1);
    const float* base = In + ((size_t)n * C + c) * Tin * VJ + v;
    Out[i] = base[(size_t)t0c * VJ] * (1.0f - w) + base[(size_t)t1c * VJ] * w;
  }
}

// -------------------------------------------------------------------------
extern "C" void kernel_launch(void* const* d_in, const int* in_sizes, int n_in,
                              void* d_out, int out_size, void* d_ws, size_t ws_size,
                              hipStream_t stream) {
  (void)in_sizes; (void)n_in; (void)out_size; (void)ws_size;

  struct Cfg { int cin, cout, stride, res, bn2, relu; };  // res: 0 none, 1 ident, 2 proj
  static const Cfg ENC[9] = {
    {3, 32, 1, 0, 1, 1}, {32, 32, 1, 1, 1, 1}, {32, 32, 2, 2, 1, 1},
    {32, 48, 1, 2, 1, 1}, {48, 48, 1, 1, 1, 1}, {48, 48, 3, 2, 1, 1},
    {48, 64, 1, 2, 1, 1}, {64, 64, 1, 1, 1, 1}, {64, 32, 1, 2, 1, 0}};
  static const Cfg DECC[5] = {
    {32, 64, 1, 2, 1, 1}, {64, 64, 1, 1, 1, 1}, {64, 48, 1, 2, 1, 1},
    {48, 48, 1, 1, 1, 1}, {48, 48, 1, 1, 1, 1}};
  const Cfg FIN = {48, 3, 1, 0, 0, 0};

  // d_in walk: setup_inputs() insertion order -> x, A, then params in dict
  // insertion order (data_bn_g, data_bn_b, enc[9], dec[5], final, ei_enc[9],
  // ei_dec[7]); each block dict in make_block insertion order with
  // conditional bn2 / res entries.
  int cur = 0;
  auto nx = [&]() { return (const float*)d_in[cur++]; };
  const float* x    = nx();
  const float* Aadj = nx();
  const float* dbg  = nx();
  const float* dbb  = nx();

  struct BP { const float *gw,*gb,*b1g,*b1b,*tw,*tb,*b2g,*b2b,*rw,*rb,*rbg,*rbb; };
  BP enc[9], dec[5], fin;
  auto grab = [&](BP& p, const Cfg& c, bool bn2) {
    p.gw = nx(); p.gb = nx(); p.b1g = nx(); p.b1b = nx(); p.tw = nx(); p.tb = nx();
    p.b2g = p.b2b = nullptr;
    if (bn2) { p.b2g = nx(); p.b2b = nx(); }
    p.rw = p.rb = p.rbg = p.rbb = nullptr;
    if (c.res == 2) { p.rw = nx(); p.rb = nx(); p.rbg = nx(); p.rbb = nx(); }
  };
  for (int i = 0; i < 9; ++i) grab(enc[i], ENC[i], true);
  for (int i = 0; i < 5; ++i) grab(dec[i], DECC[i], true);
  grab(fin, FIN, false);
  const float* ei_enc[9]; for (int i = 0; i < 9; ++i) ei_enc[i] = nx();
  const float* ei_dec[7]; for (int i = 0; i < 7; ++i) ei_dec[i] = nx();

  // Four ping-pong activation buffers; max live tensor = 512*48*12*18 floats.
  const size_t BUF = (size_t)NB * 48 * 12 * VJ;  // 5,308,416 floats = 21.2 MB
  float* b[4];
  for (int i = 0; i < 4; ++i) b[i] = (float*)d_ws + (size_t)i * BUF;

  databn_kernel<<<dim3(1296), 256, 0, stream>>>(x, dbg, dbb, b[0]);

  int curb = 0;
  int T = 12;

  auto runBlock = [&](const BP& p, const Cfg& c, const float* eip, float* outDst) {
    int Tout = (T - 1) / c.stride + 1;
    float* Xb = b[curb];
    float* Hb = b[(curb + 1) & 3];
    float* Rb = b[(curb + 2) & 3];
    float* Ob = outDst ? outDst : b[(curb + 3) & 3];

    dim3 g1((c.cout + 15) / 16, NB);
    gcn_agg_kernel<<<g1, 256, 0, stream>>>(Xb, p.gw, p.gb, Aadj, eip,
                                           p.b1g, p.b1b, Hb, c.cin, c.cout, T);
    const float* resPtr = nullptr;
    if (c.res == 1) {
      resPtr = Xb;  // identity: stride==1, cin==cout
    } else if (c.res == 2) {
      int tot = NB * c.cout * Tout * VJ;
      resproj_kernel<<<dim3((tot + 255) / 256), 256, 0, stream>>>(
          Xb, p.rw, p.rb, p.rbg, p.rbb, Rb, c.cin, c.cout, T, c.stride, Tout);
      resPtr = Rb;
    }
    dim3 g2((c.cout + 15) / 16, NB);
    tcn_kernel<<<g2, 256, 0, stream>>>(Hb, p.tw, p.tb, p.b2g, p.b2b, resPtr,
                                       Ob, c.cout, T, c.stride, Tout, c.relu);
    if (!outDst) curb = (curb + 3) & 3;
    T = Tout;
  };

  auto runUp = [&](int C, int sc) {
    int tot = NB * C * T * sc * VJ;
    float* src = b[curb];
    float* dst = b[(curb + 1) & 3];
    upsample_kernel<<<dim3((tot + 255) / 256), 256, 0, stream>>>(src, dst, C, T, sc);
    curb = (curb + 1) & 3;
    T *= sc;
  };

  // Encoder
  for (int i = 0; i < 9; ++i) runBlock(enc[i], ENC[i], ei_enc[i], nullptr);
  // Decoder: ['up3','conv','conv','conv','up2','conv','conv'] + final (ei_dec[-1])
  runUp(32, 3);
  runBlock(dec[0], DECC[0], ei_dec[1], nullptr);
  runBlock(dec[1], DECC[1], ei_dec[2], nullptr);
  runBlock(dec[2], DECC[2], ei_dec[3], nullptr);
  runUp(48, 2);
  runBlock(dec[3], DECC[3], ei_dec[5], nullptr);
  runBlock(dec[4], DECC[4], ei_dec[6], nullptr);
  runBlock(fin, FIN, ei_dec[6], (float*)d_out);
}
